// CausalSelfAttention_62259845923202
// MI455X (gfx1250) — matrix operand order repaired and checked by reading, not compiled, not run on hardware
//
#include <hip/hip_runtime.h>
#include <hip/hip_bf16.h>

typedef __bf16 bf16_t;
typedef __attribute__((ext_vector_type(4)))  __bf16 v4bf;
typedef __attribute__((ext_vector_type(8)))  __bf16 v8bf;
typedef __attribute__((ext_vector_type(16))) __bf16 v16bf;
typedef __attribute__((ext_vector_type(8)))  float  v8f;

__device__ __forceinline__ v8f v8f_zero() {
  v8f z;
#pragma unroll
  for (int i = 0; i < 8; ++i) z[i] = 0.0f;
  return z;
}

__device__ __forceinline__ v8f wmma_bf16(v16bf a, v16bf b, v8f c) {
  // D = A(16x32 bf16) * B(32x16 bf16) + C(16x16 f32)
  return __builtin_amdgcn_wmma_f32_16x16x32_bf16(
      /*neg_a=*/false, a, /*neg_b=*/false, b,
      /*c_mod=*/(short)0, c, /*reuse_a=*/false, /*reuse_b=*/false);
}

// Flat LDS addresses are {SHARED_BASE, offset[31:0]}; low 32 bits are the LDS
// byte offset the async-DMA instructions want in their VDST VGPR.
__device__ __forceinline__ unsigned lds_off(const void* p) {
  return (unsigned)(unsigned long long)p;
}

// Per-lane async DMA, global -> LDS, tracked by ASYNCcnt. INST_OFFSET is added
// to BOTH the LDS and the global address, so multiple B128 beats share one
// address pair.
__device__ __forceinline__ void async_copy32(unsigned lds_byte_off, const bf16_t* gsrc) {
  asm volatile("global_load_async_to_lds_b128 %0, %1, off\n\t"
               "global_load_async_to_lds_b128 %0, %1, off offset:16"
               :: "v"(lds_byte_off), "v"(gsrc) : "memory");
}

__device__ __forceinline__ void async_copy64(unsigned lds_byte_off, const bf16_t* gsrc) {
  asm volatile("global_load_async_to_lds_b128 %0, %1, off\n\t"
               "global_load_async_to_lds_b128 %0, %1, off offset:16\n\t"
               "global_load_async_to_lds_b128 %0, %1, off offset:32\n\t"
               "global_load_async_to_lds_b128 %0, %1, off offset:48"
               :: "v"(lds_byte_off), "v"(gsrc) : "memory");
}

__device__ __forceinline__ void wait_async0() {
  asm volatile("s_wait_asynccnt 0" ::: "memory");
}

// ---------------------------------------------------------------------------
// fp32 -> bf16 conversion (vectorized)
// ---------------------------------------------------------------------------
__global__ __launch_bounds__(256) void cvt_kernel(const float* __restrict__ in,
                                                  bf16_t* __restrict__ out, int n4) {
  int i = blockIdx.x * blockDim.x + threadIdx.x;
  if (i < n4) {
    float4 f = reinterpret_cast<const float4*>(in)[i];
    v4bf o;
    o[0] = (bf16_t)f.x; o[1] = (bf16_t)f.y; o[2] = (bf16_t)f.z; o[3] = (bf16_t)f.w;
    reinterpret_cast<v4bf*>(out)[i] = o;
  }
}

// ---------------------------------------------------------------------------
// Tiled bf16 GEMM: C[M,N] = A[M,K] * B[K,N]  (A lda=K, B ldb=N, C ldc=N)
// Block tile 128x128, BK=32, 256 threads = 8 wave32 waves, wave tile 32x64.
// Ping-pong LDS; A tile staged with GLOBAL_LOAD_ASYNC_TO_LDS_B128, B tile
// transposed through registers (2-byte granularity).
// ---------------------------------------------------------------------------
template <typename OUT_T>
__global__ __launch_bounds__(256) void gemm_kernel(const bf16_t* __restrict__ A,
                                                   const bf16_t* __restrict__ B,
                                                   OUT_T* __restrict__ C,
                                                   int K, int N) {
  constexpr int BM = 128, BN = 128, BK = 32, PAD = 8;
  constexpr int LDA_S = BK + PAD;   // 40 elems (80 B rows, 16B aligned)
  constexpr int LDB_S = BK + PAD;   // Bs stored transposed: [n][k]
  __shared__ bf16_t As[2][BM * LDA_S];
  __shared__ bf16_t Bs[2][BN * LDB_S];

  const int tid  = threadIdx.x;
  const int lane = tid & 31;
  const int wave = tid >> 5;
  const int ln   = lane & 15;
  const int half = lane >> 4;
  const int wr   = wave >> 1;   // 0..3 : 32-row band
  const int wc   = wave & 1;    // 0..1 : 64-col band
  const long row0 = (long)blockIdx.y * BM;
  const long col0 = (long)blockIdx.x * BN;

  // staging assignments
  const int ar  = tid >> 1;          // A row within tile (0..127)
  const int akc = (tid & 1) * 16;    // A k-col (0 or 16)
  const int br  = tid >> 3;          // B k-row within tile (0..31)
  const int bnc = (tid & 7) * 16;    // B n-col (0..112)
  const bf16_t* aSrc = A + (row0 + ar) * (long)K + akc;
  const bf16_t* bSrc = B + (long)br * N + col0 + bnc;

  v8f acc[2][4];
#pragma unroll
  for (int mi = 0; mi < 2; ++mi)
#pragma unroll
    for (int ni = 0; ni < 4; ++ni) acc[mi][ni] = v8f_zero();

  // ---- prologue: stage tile 0 into buffer 0 ----
  {
    async_copy32(lds_off(&As[0][ar * LDA_S + akc]), aSrc);
    v8bf blo = reinterpret_cast<const v8bf*>(bSrc)[0];
    v8bf bhi = reinterpret_cast<const v8bf*>(bSrc)[1];
#pragma unroll
    for (int i = 0; i < 8; ++i) Bs[0][(bnc + i) * LDB_S + br] = blo[i];
#pragma unroll
    for (int i = 0; i < 8; ++i) Bs[0][(bnc + 8 + i) * LDB_S + br] = bhi[i];
    wait_async0();
  }

  const int NK = K / BK;
  int cur = 0;
  for (int kt = 0; kt < NK; ++kt) {
    __syncthreads();                        // publish LDS[cur]
    const bool hasNext = (kt + 1) < NK;
    v8bf blo, bhi;
    if (hasNext) {
      const long koff = (long)(kt + 1) * BK;
      async_copy32(lds_off(&As[cur ^ 1][ar * LDA_S + akc]), aSrc + koff);
      const bf16_t* bs = bSrc + koff * N;
      blo = reinterpret_cast<const v8bf*>(bs)[0];
      bhi = reinterpret_cast<const v8bf*>(bs)[1];
      if (kt + 2 < NK)
        __builtin_prefetch(bSrc + (long)(kt + 2) * BK * N, 0, 1);
    }

    // ---- compute on LDS[cur] ----
    v16bf af[2];
#pragma unroll
    for (int mi = 0; mi < 2; ++mi) {
      const bf16_t* arow = &As[cur][(wr * 32 + mi * 16 + ln) * LDA_S];
#pragma unroll
      for (int j = 0; j < 16; ++j) {
        int k = j + 16 * half;   // contiguous k order (matches B fragments)
        af[mi][j] = arow[k];
      }
    }
#pragma unroll
    for (int ni = 0; ni < 4; ++ni) {
      const bf16_t* brow = &Bs[cur][(wc * 64 + ni * 16 + ln) * LDB_S + 16 * half];
      v16bf bfr;
      *reinterpret_cast<v8bf*>(&bfr)       = *reinterpret_cast<const v8bf*>(brow);
      *(reinterpret_cast<v8bf*>(&bfr) + 1) = *reinterpret_cast<const v8bf*>(brow + 8);
#pragma unroll
      for (int mi = 0; mi < 2; ++mi)
        acc[mi][ni] = wmma_bf16(af[mi], bfr, acc[mi][ni]);
    }

    if (hasNext) {
#pragma unroll
      for (int i = 0; i < 8; ++i) Bs[cur ^ 1][(bnc + i) * LDB_S + br] = blo[i];
#pragma unroll
      for (int i = 0; i < 8; ++i) Bs[cur ^ 1][(bnc + 8 + i) * LDB_S + br] = bhi[i];
      wait_async0();                        // our async A-chunk landed
    }
    cur ^= 1;
  }

  // ---- store (C layout: lane holds rows r + 8*half, col = ln per tile) ----
#pragma unroll
  for (int mi = 0; mi < 2; ++mi)
#pragma unroll
    for (int ni = 0; ni < 4; ++ni)
#pragma unroll
      for (int r = 0; r < 8; ++r) {
        long row = row0 + wr * 32 + mi * 16 + 8 * half + r;
        long col = col0 + wc * 64 + ni * 16 + ln;
        C[row * (long)N + col] = (OUT_T)acc[mi][ni][r];
      }
}

// ---------------------------------------------------------------------------
// Flash attention forward, causal.  qkv: (B*T, 3*H*D) bf16, out: (B*T, H*D) bf16
// Grid (Tq/128, H, B), 256 threads = 8 waves, wave owns 16 q rows.
// K tile async-DMA'd to LDS (row-major, no transpose needed for B-fragments);
// V tile transposed through VGPRs while the DMA runs in the background.
// ---------------------------------------------------------------------------
__global__ __launch_bounds__(256) void attn_fwd_kernel(const bf16_t* __restrict__ qkv,
                                                       bf16_t* __restrict__ outb) {
  constexpr int D = 128, T = 2048, RS = 6144;      // row stride of qkv
  constexpr int TQ = 128;                          // q rows per block
  constexpr int TK = 64;                           // kv rows per tile
  constexpr int LDK = D + 8;                       // Ks row pitch (136 -> 272 B)
  constexpr int LVT = TK + 8;                      // Vt/Ps row pitch (72)
  __shared__ bf16_t Ks[TK * LDK];                  // K tile row-major [kv][d]
  __shared__ bf16_t Vt[D * LVT];                   // V transposed: [d][kv]
  __shared__ bf16_t Ps[8 * 16 * LVT];              // per-wave 16x64 P tile

  const int b = blockIdx.z, h = blockIdx.y, qt = blockIdx.x;
  const int tid  = threadIdx.x;
  const int lane = tid & 31;
  const int wave = tid >> 5;
  const int ln   = lane & 15;
  const int half = lane >> 4;
  const int m0   = qt * TQ + wave * 16;            // this wave's first q row
  const long baserow = (long)b * T;
  const bf16_t* qb = qkv + baserow * RS + h * D;
  const bf16_t* kb = qb + 2048;                    // slot 1 of (3,H,D)
  const bf16_t* vb = qb + 4096;                    // slot 2
  const float scale = 0.08838834764831845f;        // 1/sqrt(128)

  // Q fragments for all 4 K-steps of D=128 (A layout), kept in registers
  v16bf qf[4];
  {
    const bf16_t* qrow = qb + (long)(m0 + ln) * RS;
#pragma unroll
    for (int f = 0; f < 4; ++f)
#pragma unroll
      for (int j = 0; j < 16; ++j) {
        int k = j + 16 * half + 32 * f;
        qf[f][j] = qrow[k];
      }
  }

  v8f o[8];
#pragma unroll
  for (int dt = 0; dt < 8; ++dt) o[dt] = v8f_zero();
  float mrow[8], lrow[8];
#pragma unroll
  for (int r = 0; r < 8; ++r) { mrow[r] = -3.0e38f; lrow[r] = 0.0f; }

  bf16_t* myP = &Ps[wave * 16 * LVT];

  // staging assignment: 256 threads, 64 kv rows, 32 d per thread
  const int skv = tid >> 2;             // 0..63
  const int sdc = (tid & 3) * 32;       // 0,32,64,96

  const int NJ = 2 * qt + 2;            // kv tiles covering rows [0, qt*128+128)
  for (int jt = 0; jt < NJ; ++jt) {
    const int j0 = jt * TK;
    const bool active = (j0 <= m0 + 15);           // any unmasked col for wave?
    __syncthreads();                               // Ks/Vt reuse guard
    // --- stage K tile via async DMA (row-major, pitch LDK) ---
    async_copy64(lds_off(&Ks[skv * LDK + sdc]),
                 kb + (long)(j0 + skv) * RS + sdc);
    // --- stage V tile transposed through VGPRs: Vt[d][kv] ---
    {
      const bf16_t* src = vb + (long)(j0 + skv) * RS + sdc;
      if (jt + 1 < NJ) {                            // prefetch next V tile
        __builtin_prefetch(src + (long)TK * RS, 0, 1);
      }
      v8bf tmp[4];
#pragma unroll
      for (int c = 0; c < 4; ++c) tmp[c] = reinterpret_cast<const v8bf*>(src)[c];
#pragma unroll
      for (int i = 0; i < 32; ++i) Vt[(sdc + i) * LVT + skv] = tmp[i >> 3][i & 7];
    }
    wait_async0();
    __syncthreads();

    float p[4][8];
    if (active) {
      // --- S = Q * K^T : 16 rows x 64 cols per wave, K frags from LDS ---
      v8f s[4];
#pragma unroll
      for (int nt = 0; nt < 4; ++nt) s[nt] = v8f_zero();
#pragma unroll
      for (int nt = 0; nt < 4; ++nt) {
        const bf16_t* krow = &Ks[(nt * 16 + ln) * LDK + 16 * half];
#pragma unroll
        for (int st = 0; st < 4; ++st) {
          v16bf kf;
          *reinterpret_cast<v8bf*>(&kf) =
              *reinterpret_cast<const v8bf*>(krow + 32 * st);
          *(reinterpret_cast<v8bf*>(&kf) + 1) =
              *reinterpret_cast<const v8bf*>(krow + 32 * st + 8);
          s[nt] = wmma_bf16(qf[st], kf, s[nt]);
        }
      }

      // --- scale, causal mask, online softmax (rows r + 8*half) ---
      const bool needMask = (j0 + TK - 1 > m0);
#pragma unroll
      for (int r = 0; r < 8; ++r) {
        const int grow = m0 + 8 * half + r;
        float mx = -3.0e38f;
#pragma unroll
        for (int nt = 0; nt < 4; ++nt) {
          float v = s[nt][r] * scale;
          if (needMask && (j0 + nt * 16 + ln) > grow) v = -3.0e38f;
          s[nt][r] = v;
          mx = fmaxf(mx, v);
        }
#pragma unroll
        for (int off = 8; off >= 1; off >>= 1)
          mx = fmaxf(mx, __shfl_xor(mx, off, 32));
        const float mnew  = fmaxf(mrow[r], mx);
        const float alpha = __expf(mrow[r] - mnew);
        float sum = 0.0f;
#pragma unroll
        for (int nt = 0; nt < 4; ++nt) {
          float e = __expf(s[nt][r] - mnew);
          p[nt][r] = e;
          sum += e;
        }
#pragma unroll
        for (int off = 8; off >= 1; off >>= 1)
          sum += __shfl_xor(sum, off, 32);
        lrow[r] = lrow[r] * alpha + sum;
        mrow[r] = mnew;
#pragma unroll
        for (int dt = 0; dt < 8; ++dt) o[dt][r] *= alpha;
      }

      // --- P (C layout) -> LDS (per-wave tile) ---
#pragma unroll
      for (int nt = 0; nt < 4; ++nt)
#pragma unroll
        for (int r = 0; r < 8; ++r)
          myP[(8 * half + r) * LVT + nt * 16 + ln] = (bf16_t)p[nt][r];
    }
    __syncthreads();

    if (active) {
      // --- P A-fragments from LDS ---
      v16bf pf[2];
#pragma unroll
      for (int st = 0; st < 2; ++st)
#pragma unroll
        for (int j = 0; j < 16; ++j) {
          int k = j + 16 * half + 32 * st;
          pf[st][j] = myP[ln * LVT + k];
        }
      // --- O += P * V ---
#pragma unroll
      for (int dt = 0; dt < 8; ++dt) {
        const bf16_t* vtrow = &Vt[(dt * 16 + ln) * LVT + 16 * half];
#pragma unroll
        for (int st = 0; st < 2; ++st) {
          v16bf vf;
          *reinterpret_cast<v8bf*>(&vf) =
              *reinterpret_cast<const v8bf*>(vtrow + 32 * st);
          *(reinterpret_cast<v8bf*>(&vf) + 1) =
              *reinterpret_cast<const v8bf*>(vtrow + 32 * st + 8);
          o[dt] = wmma_bf16(pf[st], vf, o[dt]);
        }
      }
    }
  }

  // --- normalize and store bf16 (layout (B*T, H*D)) ---
#pragma unroll
  for (int r = 0; r < 8; ++r) {
    const float inv = 1.0f / lrow[r];
    const long row = baserow + m0 + 8 * half + r;
#pragma unroll
    for (int dt = 0; dt < 8; ++dt)
      outb[row * 2048 + h * D + dt * 16 + ln] = (bf16_t)(o[dt][r] * inv);
  }
}

// ---------------------------------------------------------------------------
// Launch
// ---------------------------------------------------------------------------
extern "C" void kernel_launch(void* const* d_in, const int* in_sizes, int n_in,
                              void* d_out, int out_size, void* d_ws, size_t ws_size,
                              hipStream_t stream) {
  (void)in_sizes; (void)n_in; (void)out_size; (void)ws_size;
  const float* x      = (const float*)d_in[0];
  const float* w_qkv  = (const float*)d_in[1];
  const float* w_proj = (const float*)d_in[2];
  float* out = (float*)d_out;
  char* ws = (char*)d_ws;

  // workspace carve (bytes)
  bf16_t* xb     = (bf16_t*)(ws);                                   // 32 MiB
  bf16_t* wqkvb  = (bf16_t*)(ws + 33554432);                        // 24 MiB
  bf16_t* wprojb = (bf16_t*)(ws + 33554432 + 25165824);             //  8 MiB
  bf16_t* qkvb   = (bf16_t*)(ws + 33554432 + 25165824 + 8388608);   // 96 MiB
  bf16_t* attb   = (bf16_t*)(ws + 33554432 + 25165824 + 8388608 + 100663296); // 32 MiB

  int n4;
  n4 = 8192 * 2048 / 4; cvt_kernel<<<n4 / 256, 256, 0, stream>>>(x, xb, n4);
  n4 = 2048 * 6144 / 4; cvt_kernel<<<n4 / 256, 256, 0, stream>>>(w_qkv, wqkvb, n4);
  n4 = 2048 * 2048 / 4; cvt_kernel<<<n4 / 256, 256, 0, stream>>>(w_proj, wprojb, n4);

  // qkv = x @ w_qkv   (8192x2048 * 2048x6144 -> bf16)
  gemm_kernel<bf16_t><<<dim3(6144 / 128, 8192 / 128), 256, 0, stream>>>(
      xb, wqkvb, qkvb, 2048, 6144);

  // causal flash attention
  attn_fwd_kernel<<<dim3(2048 / 128, 16, 4), 256, 0, stream>>>(qkvb, attb);

  // out = att @ w_proj (8192x2048 * 2048x2048 -> f32)
  gemm_kernel<float><<<dim3(2048 / 128, 8192 / 128), 256, 0, stream>>>(
      attb, wprojb, out, 2048, 2048);
}
